// VerticalSelfAttention_77506979823692
// MI455X (gfx1250) — matrix operand order repaired and checked by reading, not compiled, not run on hardware
//
#include <hip/hip_runtime.h>
#include <hip/hip_bf16.h>

// ---------------------------------------------------------------------------
// VerticalSelfAttention fused transformer block for gfx1250 (MI455X, wave32).
// One workgroup per (b, c) sequence: QKV -> flash attention (bf16 WMMA,
// fp32 accum) -> out-proj -> LayerNorm -> FFN -> masked row-mean.
// Round 2: unroll control to kill scratch spills observed in round 1.
// ---------------------------------------------------------------------------

typedef __bf16 bf16_t;
typedef __attribute__((ext_vector_type(16))) bf16_t v16bf;
typedef __attribute__((ext_vector_type(8)))  float  v8f;

#define E_DIM 256
#define R_DIM 256
#define MAXC  32
#define NHEAD 4
#define HDIM  64

#define CP 264   // pitch (bf16 elems) for ctx / ln / ffn-hidden buffers
#define KP 72    // pitch for per-head K slice  [256][64+8]
#define VP 264   // pitch for per-head V^T slice [64][256+8]
#define QP 72    // pitch for per-wave Q staging [16][64+8]

// dynamic-LDS byte offsets
#define CS_OFF 0
#define LN_OFF 135168                  // 256*264*2
#define KS_OFF LN_OFF                  // K slice aliases LN region (dead then)
#define VS_OFF (LN_OFF + 36864)        // after 256*72*2
#define QS_OFF 270336                  // CS + LN
#define SMEM_BYTES 288768              // + 8*16*72*2

// workspace (bf16 weights) element offsets
#define WQKV_OFF 0
#define WOUT_OFF (768 * 256)
#define W1_OFF   (WOUT_OFF + 256 * 256)
#define W2_OFF   (W1_OFF + 256 * 256)

__device__ __forceinline__ v8f wmma_bf16(v16bf a, v16bf b, v8f c) {
  return __builtin_amdgcn_wmma_f32_16x16x32_bf16(false, a, false, b,
                                                 (short)0, c, false, false);
}

__device__ __forceinline__ v8f vzero() {
  v8f r;
#pragma unroll
  for (int j = 0; j < 8; ++j) r[j] = 0.f;
  return r;
}
__device__ __forceinline__ v8f vfill(float s) {
  v8f r;
#pragma unroll
  for (int j = 0; j < 8; ++j) r[j] = s;
  return r;
}

// two 16B chunks -> one 16-element bf16 fragment
__device__ __forceinline__ v16bf frag2(const bf16_t* p0, const bf16_t* p1) {
  union { uint4 u[2]; v16bf v; } r;
  r.u[0] = *(const uint4*)p0;
  r.u[1] = *(const uint4*)p1;
  return r.v;
}
// A-matrix (16x32, 16-bit) fragment: lane holds K {base+0..7, base+16..23};
// caller pre-offsets the pointer by +8 elems for lanes 16..31.
__device__ __forceinline__ v16bf afrag(const bf16_t* p) { return frag2(p, p + 16); }
// B-matrix (32x16, 16-bit) fragment: lane holds K {base+0..7, base+16..23};
// caller pre-offsets the pointer by +8 elems for lanes 16..31.
__device__ __forceinline__ v16bf bfrag(const bf16_t* p) { return frag2(p, p + 16); }
// A fragment straight from fp32 global memory (x), converting to bf16.
__device__ __forceinline__ v16bf afrag_f32(const float* p) {
  v16bf r;
#pragma unroll
  for (int i = 0; i < 8; ++i) r[i] = (bf16_t)p[i];
#pragma unroll
  for (int i = 0; i < 8; ++i) r[i + 8] = (bf16_t)p[i + 16];
  return r;
}

__global__ void convert_f32_bf16(const float* __restrict__ s,
                                 bf16_t* __restrict__ d, int n) {
  int i = blockIdx.x * 256 + threadIdx.x;
  if (i < n) d[i] = (bf16_t)s[i];
}

__global__ void __launch_bounds__(256, 1)
vsa_main(const float* __restrict__ x,
         const int* __restrict__ real_cols,
         const int* __restrict__ real_rows,
         const bf16_t* __restrict__ wsW,
         const float* __restrict__ b_qkv,
         const float* __restrict__ b_out,
         const float* __restrict__ ln_g,
         const float* __restrict__ ln_b,
         const float* __restrict__ b1v,
         const float* __restrict__ b2v,
         float* __restrict__ out) {
  extern __shared__ char smem[];
  bf16_t* cs  = (bf16_t*)(smem + CS_OFF);   // ctx bf16, later FFN hidden
  bf16_t* lnb = (bf16_t*)(smem + LN_OFF);   // LN output (after attention)
  bf16_t* ks  = (bf16_t*)(smem + KS_OFF);   // per-head K slice (pre-LN phase)
  bf16_t* vs  = (bf16_t*)(smem + VS_OFF);   // per-head V^T slice
  const int tid  = threadIdx.x;
  const int w    = tid >> 5;
  const int lane = tid & 31;
  const int l16  = lane & 15;
  const int hi   = lane >> 4;
  bf16_t* qst = (bf16_t*)(smem + QS_OFF) + w * (16 * QP);

  const int bc = blockIdx.x;
  const int b  = bc >> 5;     // maxC == 32
  const int c  = bc & 31;
  const int nrows = real_rows[b];

  if (c >= real_cols[b]) {    // padded column: zero output, uniform exit
    out[(size_t)bc * E_DIM + tid] = 0.f;
    return;
  }

  const float*  xb   = x + (size_t)bc * (R_DIM * E_DIM);
  const bf16_t* Wqkv = wsW + WQKV_OFF;

  // ======================= per-head K/V + flash attention ==================
#pragma unroll 1
  for (int h = 0; h < NHEAD; ++h) {
    // ---- K and V slices for this head (wave owns 2 row-tiles) ----
#pragma unroll 1
    for (int sti = 0; sti < 2; ++sti) {
      const int st = w * 2 + sti;
      const float* xr = xb + (size_t)(st * 16 + l16) * E_DIM + hi * 8;
      __builtin_prefetch(xr + E_DIM * 16, 0, 1);  // next row-tile of x
      v16bf xA[8];
#pragma unroll
      for (int kk = 0; kk < 8; ++kk) xA[kk] = afrag_f32(xr + kk * 32);
      // K features: 256 + h*64 ...
#pragma unroll 1
      for (int ft = 0; ft < 4; ++ft) {
        const int f = 256 + h * 64 + ft * 16 + l16;
        v8f cacc = vfill(b_qkv[f]);
        const bf16_t* wr = Wqkv + (size_t)f * E_DIM + hi * 8;
#pragma unroll
        for (int kk = 0; kk < 8; ++kk)
          cacc = wmma_bf16(xA[kk], bfrag(wr + kk * 32), cacc);
#pragma unroll
        for (int j = 0; j < 8; ++j)
          ks[(st * 16 + j + hi * 8) * KP + ft * 16 + l16] = (bf16_t)cacc[j];
      }
      // V features: 512 + h*64 ... stored transposed vs[d][s]
#pragma unroll 1
      for (int ft = 0; ft < 4; ++ft) {
        const int f = 512 + h * 64 + ft * 16 + l16;
        v8f cacc = vfill(b_qkv[f]);
        const bf16_t* wr = Wqkv + (size_t)f * E_DIM + hi * 8;
#pragma unroll
        for (int kk = 0; kk < 8; ++kk)
          cacc = wmma_bf16(xA[kk], bfrag(wr + kk * 32), cacc);
#pragma unroll
        for (int j = 0; j < 8; ++j)
          vs[(ft * 16 + l16) * VP + st * 16 + j + hi * 8] = (bf16_t)cacc[j];
      }
    }
    __syncthreads();

    // ---- flash attention, wave owns 2 query tiles ----
#pragma unroll 1
    for (int qi = 0; qi < 2; ++qi) {
      const int qt = w * 2 + qi;
      {  // compute Q tile for (qt, h) -> per-wave staging (scale folded in)
        const float* xr = xb + (size_t)(qt * 16 + l16) * E_DIM + hi * 8;
        v16bf xA[8];
#pragma unroll
        for (int kk = 0; kk < 8; ++kk) xA[kk] = afrag_f32(xr + kk * 32);
#pragma unroll 1
        for (int ft = 0; ft < 4; ++ft) {
          const int f = h * 64 + ft * 16 + l16;
          v8f cacc = vfill(b_qkv[f]);
          const bf16_t* wr = Wqkv + (size_t)f * E_DIM + hi * 8;
#pragma unroll
          for (int kk = 0; kk < 8; ++kk)
            cacc = wmma_bf16(xA[kk], bfrag(wr + kk * 32), cacc);
#pragma unroll
          for (int j = 0; j < 8; ++j)
            qst[(j + hi * 8) * QP + ft * 16 + l16] = (bf16_t)(cacc[j] * 0.125f);
        }
      }
      asm volatile("s_wait_dscnt 0x0" ::: "memory");  // wave-local LDS ordering
      const v16bf qB0 = bfrag(qst + l16 * QP + hi * 8);
      const v16bf qB1 = bfrag(qst + l16 * QP + 32 + hi * 8);

      float mrun = -3.0e38f, lrun = 0.f;
      v8f ctx0 = vzero(), ctx1 = vzero(), ctx2 = vzero(), ctx3 = vzero();

#pragma unroll 1
      for (int kb = 0; kb < 8; ++kb) {  // 32 keys per step
        const bf16_t* kr0 = ks + ((2 * kb) * 16 + l16) * KP + hi * 8;
        const bf16_t* kr1 = ks + ((2 * kb + 1) * 16 + l16) * KP + hi * 8;
        v8f s0 = vzero(), s1 = vzero();
        s0 = wmma_bf16(afrag(kr0), qB0, s0);
        s0 = wmma_bf16(afrag(kr0 + 32), qB1, s0);
        s1 = wmma_bf16(afrag(kr1), qB0, s1);
        s1 = wmma_bf16(afrag(kr1 + 32), qB1, s1);

        const int sb0 = kb * 32 + hi * 8;
        const int sb1 = kb * 32 + 16 + hi * 8;
        float tm = -3.0e38f;
#pragma unroll
        for (int j = 0; j < 8; ++j) {
          if (sb0 + j >= nrows) s0[j] = -1.0e9f;
          if (sb1 + j >= nrows) s1[j] = -1.0e9f;
          tm = fmaxf(tm, fmaxf(s0[j], s1[j]));
        }
        tm = fmaxf(tm, __shfl_xor(tm, 16));
        const float mnew = fmaxf(mrun, tm);
        const float corr = __expf(mrun - mnew);
        float ps = 0.f;
        v16bf pA;  // exp(scores) is already in A-fragment layout
#pragma unroll
        for (int j = 0; j < 8; ++j) {
          const float p0 = __expf(s0[j] - mnew);
          const float p1 = __expf(s1[j] - mnew);
          ps += p0 + p1;
          pA[j]     = (bf16_t)p0;
          pA[j + 8] = (bf16_t)p1;
        }
        ps += __shfl_xor(ps, 16);
        lrun = lrun * corr + ps;
        mrun = mnew;
#pragma unroll
        for (int j = 0; j < 8; ++j) {  // broadcast per-row correction
          const float fj = __shfl(corr, j + hi * 8);
          ctx0[j] *= fj; ctx1[j] *= fj; ctx2[j] *= fj; ctx3[j] *= fj;
        }
        const bf16_t* vr = vs + l16 * VP + kb * 32 + hi * 8;
        ctx0 = wmma_bf16(pA, bfrag(vr), ctx0);
        ctx1 = wmma_bf16(pA, bfrag(vr + 16 * VP), ctx1);
        ctx2 = wmma_bf16(pA, bfrag(vr + 32 * VP), ctx2);
        ctx3 = wmma_bf16(pA, bfrag(vr + 48 * VP), ctx3);
      }
      const float inv = 1.f / lrun;
#pragma unroll
      for (int j = 0; j < 8; ++j) {
        const float fj = __shfl(inv, j + hi * 8);
        bf16_t* cr = cs + (size_t)(qt * 16 + j + hi * 8) * CP + h * 64 + l16;
        cr[0]  = (bf16_t)(ctx0[j] * fj);
        cr[16] = (bf16_t)(ctx1[j] * fj);
        cr[32] = (bf16_t)(ctx2[j] * fj);
        cr[48] = (bf16_t)(ctx3[j] * fj);
      }
    }
    __syncthreads();  // K/V slice region recycled next head / LN phase
  }

  // ======================= out-projection + LayerNorm ======================
  const bf16_t* Wout = wsW + WOUT_OFF;
#pragma unroll 1
  for (int ri = 0; ri < 2; ++ri) {
    const int rt = w * 2 + ri;
    v16bf aA[8];
    const bf16_t* ar = cs + (size_t)(rt * 16 + l16) * CP + hi * 8;
#pragma unroll
    for (int kk = 0; kk < 8; ++kk) aA[kk] = afrag(ar + kk * 32);
    v8f ssum = vzero(), ssq = vzero();
#pragma unroll 1
    for (int ft = 0; ft < 16; ++ft) {  // stats pass
      v8f cacc = vfill(b_out[ft * 16 + l16]);
      const bf16_t* wr = Wout + (size_t)(ft * 16 + l16) * E_DIM + hi * 8;
#pragma unroll
      for (int kk = 0; kk < 8; ++kk)
        cacc = wmma_bf16(aA[kk], bfrag(wr + kk * 32), cacc);
#pragma unroll
      for (int j = 0; j < 8; ++j) { ssum[j] += cacc[j]; ssq[j] += cacc[j] * cacc[j]; }
    }
    float mu[8], rs[8];
#pragma unroll
    for (int j = 0; j < 8; ++j) {
      float s = ssum[j], q = ssq[j];
#pragma unroll
      for (int off = 1; off < 16; off <<= 1) {
        s += __shfl_xor(s, off);
        q += __shfl_xor(q, off);
      }
      const float m = s * (1.f / E_DIM);
      mu[j] = m;
      rs[j] = rsqrtf(q * (1.f / E_DIM) - m * m + 1e-5f);
    }
#pragma unroll 1
    for (int ft = 0; ft < 16; ++ft) {  // normalize pass (recompute)
      const int f = ft * 16 + l16;
      const float g = ln_g[f], bb = ln_b[f];
      v8f cacc = vfill(b_out[f]);
      const bf16_t* wr = Wout + (size_t)f * E_DIM + hi * 8;
#pragma unroll
      for (int kk = 0; kk < 8; ++kk)
        cacc = wmma_bf16(aA[kk], bfrag(wr + kk * 32), cacc);
#pragma unroll
      for (int j = 0; j < 8; ++j)
        lnb[(size_t)(rt * 16 + j + hi * 8) * CP + f] =
            (bf16_t)((cacc[j] - mu[j]) * rs[j] * g + bb);
    }
  }
  __syncthreads();

  // ======================= FFN1 (ReLU) =====================================
  const bf16_t* W1 = wsW + W1_OFF;
#pragma unroll 1
  for (int ri = 0; ri < 2; ++ri) {
    const int rt = w * 2 + ri;
    v16bf aA[8];
    const bf16_t* ar = lnb + (size_t)(rt * 16 + l16) * CP + hi * 8;
#pragma unroll
    for (int kk = 0; kk < 8; ++kk) aA[kk] = afrag(ar + kk * 32);
#pragma unroll 1
    for (int ft = 0; ft < 16; ++ft) {
      const int f = ft * 16 + l16;
      v8f cacc = vfill(b1v[f]);
      const bf16_t* wr = W1 + (size_t)f * E_DIM + hi * 8;
#pragma unroll
      for (int kk = 0; kk < 8; ++kk)
        cacc = wmma_bf16(aA[kk], bfrag(wr + kk * 32), cacc);
#pragma unroll
      for (int j = 0; j < 8; ++j)  // hidden reuses cs region
        cs[(size_t)(rt * 16 + j + hi * 8) * CP + f] = (bf16_t)fmaxf(cacc[j], 0.f);
    }
  }
  __syncthreads();

  // ======================= FFN2 + residual + masked row-mean ===============
  const bf16_t* W2 = wsW + W2_OFF;
  const float invn = 1.f / (float)nrows;
#pragma unroll 1
  for (int fi = 0; fi < 2; ++fi) {
    const int ft = w * 2 + fi;
    const int f  = ft * 16 + l16;
    const float bias = b2v[f];
    const bf16_t* wr = W2 + (size_t)f * E_DIM + hi * 8;
    v16bf wB[8];
#pragma unroll
    for (int kk = 0; kk < 8; ++kk) wB[kk] = bfrag(wr + kk * 32);
    float racc = 0.f;
#pragma unroll 1
    for (int rt = 0; rt < 16; ++rt) {
      v8f cacc = vfill(bias);
      const bf16_t* ar = cs + (size_t)(rt * 16 + l16) * CP + hi * 8;
#pragma unroll
      for (int kk = 0; kk < 8; ++kk)
        cacc = wmma_bf16(afrag(ar + kk * 32), wB[kk], cacc);
#pragma unroll
      for (int j = 0; j < 8; ++j) {
        const int row = rt * 16 + j + hi * 8;
        const float v = cacc[j] + (float)lnb[(size_t)row * CP + f];
        racc += (row < nrows) ? v : 0.f;
      }
    }
    racc += __shfl_xor(racc, 16);
    if (hi == 0) out[(size_t)bc * E_DIM + f] = racc * invn;
  }
}

extern "C" void kernel_launch(void* const* d_in, const int* in_sizes, int n_in,
                              void* d_out, int out_size, void* d_ws,
                              size_t ws_size, hipStream_t stream) {
  const float* x    = (const float*)d_in[0];
  const int*   rc   = (const int*)d_in[1];
  const int*   rr   = (const int*)d_in[2];
  const float* wqkv = (const float*)d_in[3];
  const float* bqkv = (const float*)d_in[4];
  const float* wout = (const float*)d_in[5];
  const float* bout = (const float*)d_in[6];
  const float* lng  = (const float*)d_in[7];
  const float* lnbp = (const float*)d_in[8];
  const float* w1   = (const float*)d_in[9];
  const float* b1   = (const float*)d_in[10];
  const float* w2   = (const float*)d_in[11];
  const float* b2   = (const float*)d_in[12];
  bf16_t* wsW = (bf16_t*)d_ws;

  convert_f32_bf16<<<(768 * 256 + 255) / 256, 256, 0, stream>>>(wqkv, wsW + WQKV_OFF, 768 * 256);
  convert_f32_bf16<<<(256 * 256 + 255) / 256, 256, 0, stream>>>(wout, wsW + WOUT_OFF, 256 * 256);
  convert_f32_bf16<<<(256 * 256 + 255) / 256, 256, 0, stream>>>(w1,   wsW + W1_OFF,   256 * 256);
  convert_f32_bf16<<<(256 * 256 + 255) / 256, 256, 0, stream>>>(w2,   wsW + W2_OFF,   256 * 256);

  hipFuncSetAttribute((const void*)vsa_main,
                      hipFuncAttributeMaxDynamicSharedMemorySize, SMEM_BYTES);
  vsa_main<<<16 * MAXC, 256, SMEM_BYTES, stream>>>(
      x, rc, rr, wsW, bqkv, bout, lng, lnbp, b1, b2, (float*)d_out);
}
